// CausalAttention_42623255446056
// MI455X (gfx1250) — matrix operand order repaired and checked by reading, not compiled, not run on hardware
//
#include <hip/hip_runtime.h>

typedef __attribute__((ext_vector_type(16))) _Float16 v16h;
typedef __attribute__((ext_vector_type(8)))  _Float16 v8h;
typedef __attribute__((ext_vector_type(8)))  float    v8f;

static constexpr int kDIn  = 256;
static constexpr int kDOut = 128;
static constexpr int kSeq  = 4096;
static constexpr int kBat  = 4;
static constexpr size_t kQKV = (size_t)kBat * kSeq * kDOut;  // 2M elements

// Hardware exp2: single v_exp_f32 (TRANS op, co-executes with VALU).
__device__ __forceinline__ float fast_exp2(float x) {
  return __builtin_amdgcn_exp2f(x);
}

// ---- WMMA fragment layout helpers (CDNA5 ISA §7.12.2, wave32) ----

// A fragment, this file's k order (B matches): lanes 0-15 hold row M=lane with K = 0..15,
// lanes 16-31 hold row M=lane-16 with K = 16..31, packed in pairs.
__device__ __forceinline__ int a_koff(int v, int half) {
  return 2 * v + 16 * half;
}

__device__ __forceinline__ v16h load_a_from_f32(const float* __restrict__ A, int ld,
                                                int mbase, int kbase, int lane) {
  const int half = lane >> 4, r = lane & 15;
  const float* row = A + (size_t)(mbase + r) * ld + kbase;
  v16h f;
#pragma unroll
  for (int v = 0; v < 8; ++v) {
    const int k = a_koff(v, half);
    f[2 * v]     = (_Float16)row[k];      // pairs are contiguous -> b64 loads
    f[2 * v + 1] = (_Float16)row[k + 1];
  }
  return f;
}

__device__ __forceinline__ v16h load_a_from_h(const _Float16* __restrict__ A, int ld,
                                              int mbase, int kbase, int lane) {
  const int half = lane >> 4, r = lane & 15;
  const _Float16* row = A + (size_t)(mbase + r) * ld + kbase;
  v16h f;
#pragma unroll
  for (int v = 0; v < 8; ++v) {
    const int k = a_koff(v, half);
    f[2 * v]     = row[k];
    f[2 * v + 1] = row[k + 1];
  }
  return f;
}

// B fragment for logical B = K^T: element(k,n) = K[nbase+n][kbase+k]
// (K row-major [S, d]: each lane's 16 k-values are 32 contiguous bytes -> b128s)
__device__ __forceinline__ v16h load_bT_from_h(const _Float16* __restrict__ K, int ld,
                                               int nbase, int kbase, int lane) {
  const int half = lane >> 4;
  const _Float16* row = K + (size_t)(nbase + (lane & 15)) * ld + kbase + 16 * half;
  return *(const v16h*)row;
}

// ---- Kernel 0: pre-pack W (fp32 col walks, once) into fragment-linear f16 ----
__global__ __launch_bounds__(32) void pack_w(
    const float* __restrict__ Wq, const float* __restrict__ Wk,
    const float* __restrict__ Wv, _Float16* __restrict__ wpack) {
  const int lane = threadIdx.x & 31;
  const int tile = blockIdx.x;                 // [0, 192): which*64 + ks*8 + nt
  const int which = tile >> 6, ks = (tile >> 3) & 7, nt = tile & 7;
  const float* W = (which == 0) ? Wq : (which == 1) ? Wk : Wv;
  const int half = lane >> 4, nc = lane & 15;
  v16h f;
#pragma unroll
  for (int j = 0; j < 16; ++j)
    f[j] = (_Float16)W[(size_t)(ks * 32 + 16 * half + j) * kDOut + nt * 16 + nc];
  *(v16h*)(wpack + (size_t)tile * 512 + (size_t)lane * 16) = f;
}

// ---- Kernel 1: QKV projection. q/k row-major f16; V in B-fragment-linear ----
__global__ __launch_bounds__(128) void qkv_proj_wmma(
    const float* __restrict__ x, const _Float16* __restrict__ wpack,
    _Float16* __restrict__ qh, _Float16* __restrict__ kh,
    _Float16* __restrict__ vpack) {
  const int lane = threadIdx.x & 31;
  const int wave = threadIdx.x >> 5;
  const int mtile = blockIdx.x * 4 + wave;  // [0, 1024)
  const int which = blockIdx.y;             // 0:q 1:k 2:v
  const int mbase = mtile * 16;

  v16h a[8];
#pragma unroll
  for (int ks = 0; ks < 8; ++ks)
    a[ks] = load_a_from_f32(x, kDIn, mbase, ks * 32, lane);

  const int half = lane >> 4, nc = lane & 15;
#pragma unroll
  for (int nt = 0; nt < 8; ++nt) {
    v8f c = {};
#pragma unroll
    for (int ks = 0; ks < 8; ++ks) {
      const v16h b = *(const v16h*)(wpack +
          (size_t)((which * 8 + ks) * 8 + nt) * 512 + (size_t)lane * 16);
      c = __builtin_amdgcn_wmma_f32_16x16x32_f16(false, a[ks], false, b,
                                                 (short)0, c, false, false);
    }
    if (which < 2) {
      _Float16* out = (which == 0) ? qh : kh;
#pragma unroll
      for (int v = 0; v < 8; ++v)
        out[(size_t)(mbase + v + 8 * half) * kDOut + nt * 16 + nc] = (_Float16)c[v];
    } else {
      // C lane (half, nc) holds packed-lane (halfB, nc) positions j = 8*half+v.
      v8h pk;
#pragma unroll
      for (int v = 0; v < 8; ++v) pk[v] = (_Float16)c[v];
      const int block = mbase >> 5, halfB = (mbase >> 4) & 1;
      _Float16* dst = vpack + ((size_t)(block * 8 + nt) * 32 +
                               halfB * 16 + nc) * 16 + 8 * half;
      *(v8h*)dst = pk;
    }
  }
}

// ------------- Kernel 2: causal softmax attention, one wave / 16 queries ----

__global__ __launch_bounds__(128) void causal_attn_wmma(
    const _Float16* __restrict__ qh, const _Float16* __restrict__ kh,
    const _Float16* __restrict__ vpack, float* __restrict__ ctx,
    float* __restrict__ attn) {
  const int lane = threadIdx.x & 31;
  const int wave = threadIdx.x >> 5;
  const int qt = blockIdx.x * 4 + wave;  // query tile [0,256)
  const int b  = blockIdx.y;
  const int qbase = qt * 16;

  const _Float16* Q = qh + (size_t)b * kSeq * kDOut;
  const _Float16* K = kh + (size_t)b * kSeq * kDOut;
  float* attnB = attn + (size_t)b * kSeq * kSeq;

  __shared__ float lds[4][16 * 32];  // per-wave P-tile scratch
  float* sc = lds[wave];

  v16h qf[4];
#pragma unroll
  for (int ks = 0; ks < 4; ++ks)
    qf[ks] = load_a_from_h(Q, kDOut, qbase, ks * 32, lane);

  const int half = lane >> 4, nc = lane & 15;
  // base-2 softmax: exp2(x * (log2(e)/sqrt(128))) == exp(x / sqrt(128))
  const float scale2 = 0.12753553519471536f;  // 1.4426950408889634 / sqrt(128)

  // Per-lane online max/sum (no cross-lane traffic inside the sweep).
  float mrow[8], lrow[8];
#pragma unroll
  for (int v = 0; v < 8; ++v) { mrow[v] = -1e30f; lrow[v] = 0.f; }

  // ---- Sweep 1: per-lane online softmax stats ----
  for (int jt = 0; jt <= qt; ++jt) {
    const int kb = jt * 16;
    if (jt < qt)
      __builtin_prefetch(K + (size_t)(kb + 16) * kDOut, 0, 0);
    v8f s = {};
#pragma unroll
    for (int ks = 0; ks < 4; ++ks) {
      const v16h bk = load_bT_from_h(K, kDOut, kb, ks * 32, lane);
      s = __builtin_amdgcn_wmma_f32_16x16x32_f16(false, qf[ks], false, bk,
                                                 (short)0, s, false, false);
    }
#pragma unroll
    for (int v = 0; v < 8; ++v) {
      const int qi = qbase + v + 8 * half;
      const int ki = kb + nc;
      const float val = (ki <= qi) ? s[v] * scale2 : -1e30f;
      const float mn = fmaxf(mrow[v], val);
      lrow[v] = lrow[v] * fast_exp2(mrow[v] - mn) + fast_exp2(val - mn);
      mrow[v] = mn;
    }
  }

  // ---- One cross-lane log-sum-exp merge per row (4 butterfly steps) ----
  float linv[8];
#pragma unroll
  for (int v = 0; v < 8; ++v) {
    float m = mrow[v], l = lrow[v];
#pragma unroll
    for (int off = 1; off < 16; off <<= 1) {  // stays inside 16-lane half
      const float mo = __shfl_xor(m, off, 32);
      const float lo = __shfl_xor(l, off, 32);
      const float mn = fmaxf(m, mo);
      l = l * fast_exp2(m - mn) + lo * fast_exp2(mo - mn);
      m = mn;
    }
    mrow[v] = m;
    linv[v] = 1.0f / l;
  }

  // ---- Sweep 2: final probabilities (attn written once) + context = P@V ----
  v8f o[8] = {};
  for (int t = 0; 2 * t <= qt; ++t) {
    const int kb32 = t * 32;
#pragma unroll
    for (int sub = 0; sub < 2; ++sub) {
      const int jt = 2 * t + sub;
      const int kb = kb32 + sub * 16;
      if (jt <= qt) {
        v8f s = {};
#pragma unroll
        for (int ks = 0; ks < 4; ++ks) {
          const v16h bk = load_bT_from_h(K, kDOut, kb, ks * 32, lane);
          s = __builtin_amdgcn_wmma_f32_16x16x32_f16(false, qf[ks], false, bk,
                                                     (short)0, s, false, false);
        }
#pragma unroll
        for (int v = 0; v < 8; ++v) {
          const int m = v + 8 * half;
          const float val = (kb + nc <= qbase + m) ? s[v] * scale2 : -1e30f;
          sc[m * 32 + sub * 16 + nc] = fast_exp2(val - mrow[v]) * linv[v];
        }
      } else {
#pragma unroll
        for (int v = 0; v < 8; ++v)
          sc[(v + 8 * half) * 32 + sub * 16 + nc] = 0.f;
      }
    }
    // Same-wave DS ops are in-order; wait + compiler fence before re-reading.
    asm volatile("s_wait_dscnt 0" ::: "memory");

    // C->A transpose read-back for the P fragment (16x32 f16)
    v16h pf;
    {
      const int r = lane & 15;
#pragma unroll
      for (int v = 0; v < 8; ++v) {
        const int k = a_koff(v, half);
        pf[2 * v]     = (_Float16)sc[r * 32 + k];
        pf[2 * v + 1] = (_Float16)sc[r * 32 + k + 1];
      }
    }

    // Coalesced attn tile store from LDS: 16 rows x 32 cols as float4s.
#pragma unroll
    for (int it = 0; it < 4; ++it) {
      const int chunk = it * 32 + lane;
      const int row = chunk >> 3, c4 = chunk & 7;
      const float4 pv = *(const float4*)(sc + row * 32 + c4 * 4);
      *(float4*)(attnB + (size_t)(qbase + row) * kSeq + kb32 + c4 * 4) = pv;
    }

    // context += P (16x32) @ V (32x128), V pre-packed fragment-linear
    const _Float16* vp = vpack +
        (size_t)(b * (kSeq / 32) + (kb32 >> 5)) * 8 * 512;
#pragma unroll
    for (int nt = 0; nt < 8; ++nt) {
      const v16h bv = *(const v16h*)(vp + (size_t)nt * 512 + (size_t)lane * 16);
      o[nt] = __builtin_amdgcn_wmma_f32_16x16x32_f16(false, pf, false, bv,
                                                     (short)0, o[nt], false, false);
    }
  }

  // ---- Store context [B, S, 128] fp32 ----
#pragma unroll
  for (int nt = 0; nt < 8; ++nt) {
#pragma unroll
    for (int v = 0; v < 8; ++v) {
      const int qi = qbase + v + 8 * half;
      ctx[((size_t)b * kSeq + qi) * kDOut + nt * 16 + nc] = o[nt][v];
    }
  }

  // ---- Zero-fill remaining fully-masked columns (32-key aligned start) ----
  const int zstart = 32 * ((qt >> 1) + 1);
  const float4 z4 = make_float4(0.f, 0.f, 0.f, 0.f);
  for (int r = 0; r < 16; ++r) {
    float* row = attnB + (size_t)(qbase + r) * kSeq;
    for (int c = zstart + lane * 4; c < kSeq; c += 32 * 4)
      *(float4*)(row + c) = z4;
  }
}

// ---------------------------------------------------------------------------

extern "C" void kernel_launch(void* const* d_in, const int* in_sizes, int n_in,
                              void* d_out, int out_size, void* d_ws, size_t ws_size,
                              hipStream_t stream) {
  const float* x  = (const float*)d_in[0];
  const float* Wq = (const float*)d_in[1];
  const float* Wk = (const float*)d_in[2];
  const float* Wv = (const float*)d_in[3];

  float* ctx  = (float*)d_out;
  float* attn = (float*)d_out + (size_t)kBat * kSeq * kDOut;

  _Float16* qh    = (_Float16*)d_ws;       // 4 MB
  _Float16* kh    = qh + kQKV;             // 4 MB
  _Float16* vpack = kh + kQKV;             // 4 MB (fragment-linear V)
  _Float16* wpack = vpack + kQKV;          // 192 KB (fragment-linear W)

  pack_w<<<192, 32, 0, stream>>>(Wq, Wk, Wv, wpack);
  qkv_proj_wmma<<<dim3(256, 3), 128, 0, stream>>>(x, wpack, qh, kh, vpack);
  causal_attn_wmma<<<dim3(64, kBat), 128, 0, stream>>>(qh, kh, vpack, ctx, attn);
}
